// ChebConvWithAttStatic_21182778704671
// MI455X (gfx1250) — compile-verified, matrix operand order checked by reading, not run on hardware
//
#include <hip/hip_runtime.h>

typedef __attribute__((ext_vector_type(16))) __bf16 v16bf;
typedef __attribute__((ext_vector_type(8)))  __bf16 bf8v;
typedef __attribute__((ext_vector_type(8)))  float  v8f;
typedef __attribute__((ext_vector_type(4)))  float  f4v;

#define B_ 16
#define T_ 128
#define V_ 256
#define F_ 64
#define K_ 3
#define O_ 64
#define KV_ (K_ * V_)   // 768: total contraction length of GEMM2

// Native f32 -> bf16 conversion: let the compiler select v_cvt_pk_bf16_f32 /
// v_cvt_bf16_f32 on gfx1250 instead of a 3-op manual RNE sequence.
__device__ __forceinline__ __bf16 f2bf(float f) {
    return static_cast<__bf16>(f);
}

__global__ __launch_bounds__(256)
void cheb_att_fused(const float* __restrict__ x,
                    const float* __restrict__ Att,
                    const float* __restrict__ cheb,
                    const float* __restrict__ Theta,
                    float* __restrict__ out)
{
    // zT[o][kk], kk = k*256+u : transposed intermediate, bf16 (96 KB)
    __shared__ __bf16 zT[O_ * KV_];
    // thT[k][o][f] : Theta transposed per k, bf16 (24 KB)
    __shared__ __bf16 thT[K_ * O_ * F_];
    // mt[v][uu] : (cheb*Att)^T chunk, 32 u's per pass, bf16 (16 KB)
    __shared__ __bf16 mt[V_ * 32];

    const int bt   = blockIdx.x;
    const int b    = bt / T_;
    const int t    = bt % T_;
    const int tid  = threadIdx.x;
    const int wave = tid >> 5;
    const int lane = tid & 31;
    const int ln   = lane & 15;   // M (A) / N (B) index within tile
    const int hi   = lane >> 4;   // selects K-slot half

    // ---- stage Theta^T into LDS (coalesced over o) ----
    for (int idx = tid; idx < K_ * F_ * O_; idx += 256) {
        int k   = idx / (F_ * O_);
        int rem = idx - k * (F_ * O_);
        int f   = rem / O_;
        int o   = rem - f * O_;
        thT[(k * O_ + o) * F_ + f] = f2bf(Theta[idx]);
    }
    __syncthreads();

    // ---- GEMM1: z[(k,u)][o] = sum_f x[b,t,u,f] * Theta[k,f,o]; write zT ----
    const float* xbt = x + (size_t)(b * T_ + t) * V_ * F_;
    for (int i = 0; i < 6; ++i) {
        const int rt = wave * 6 + i;      // row tile 0..47 of z
        const int k  = rt >> 4;
        const int ut = rt & 15;

        // A fragments (x rows, f32->bf16), for K chunks fc=0 and fc=32
        v16bf A[2];
        const float* xrow = xbt + (ut * 16 + ln) * F_;
        #pragma unroll
        for (int c = 0; c < 2; ++c) {
            const int k0 = c * 32 + hi * 8;
            f4v a0 = *(const f4v*)(xrow + k0);
            f4v a1 = *(const f4v*)(xrow + k0 + 4);
            f4v b0 = *(const f4v*)(xrow + k0 + 16);
            f4v b1 = *(const f4v*)(xrow + k0 + 20);
            union { v16bf v; __bf16 e[16]; } ua;
            #pragma unroll
            for (int j = 0; j < 4; ++j) {
                ua.e[j]      = f2bf(a0[j]);
                ua.e[4 + j]  = f2bf(a1[j]);
                ua.e[8 + j]  = f2bf(b0[j]);
                ua.e[12 + j] = f2bf(b1[j]);
            }
            A[c] = ua.v;
        }

        #pragma unroll
        for (int ct = 0; ct < 4; ++ct) {
            v8f acc = {};
            #pragma unroll
            for (int c = 0; c < 2; ++c) {
                const __bf16* bp = &thT[(k * O_ + ct * 16 + ln) * F_ + c * 32 + hi * 8];
                union { v16bf v; bf8v h[2]; } ub;
                ub.h[0] = *(const bf8v*)(bp);
                ub.h[1] = *(const bf8v*)(bp + 16);
                acc = __builtin_amdgcn_wmma_f32_16x16x32_bf16(
                          false, A[c], false, ub.v, (short)0, acc, false, false);
            }
            // transposed store: lane's 8 rows are contiguous in (k,u) -> one b128 DS store
            bf8v pack;
            #pragma unroll
            for (int r = 0; r < 8; ++r) pack[r] = f2bf(acc[r]);
            *(bf8v*)(&zT[(ct * 16 + ln) * KV_ + rt * 16 + hi * 8]) = pack;
        }
    }

    // ---- GEMM2: out[v][o] = sum_{k,u} (cheb*Att)[k,u,v] * z[(k,u)][o] ----
    v8f acc[2][4] = {};   // 2 v-tiles x 4 o-tiles per wave

    const float* Attb = Att + (size_t)b * V_ * V_;
    for (int k = 0; k < K_; ++k) {
        const float* chk = cheb + (size_t)k * V_ * V_;
        for (int uc = 0; uc < 8; ++uc) {
            __syncthreads();   // previous mt consumed (also orders zT writes on first pass)
            {  // stage mt[v][uu] = cheb[k][u][v] * Att[b][u][v], coalesced over v=tid
                const int v = tid;
                #pragma unroll 4
                for (int uu = 0; uu < 32; ++uu) {
                    const int u = uc * 32 + uu;
                    mt[v * 32 + uu] = f2bf(chk[u * V_ + v] * Attb[u * V_ + v]);
                }
            }
            __syncthreads();

            const int kk = k * V_ + uc * 32;   // column base in zT
            v16bf Af[2];
            #pragma unroll
            for (int a = 0; a < 2; ++a) {
                const __bf16* ap = &mt[((wave * 2 + a) * 16 + ln) * 32 + hi * 8];
                union { v16bf v; bf8v h[2]; } ua;
                ua.h[0] = *(const bf8v*)(ap);
                ua.h[1] = *(const bf8v*)(ap + 16);
                Af[a] = ua.v;
            }
            #pragma unroll
            for (int ct = 0; ct < 4; ++ct) {
                const __bf16* bp = &zT[(ct * 16 + ln) * KV_ + kk + hi * 8];
                union { v16bf v; bf8v h[2]; } ub;
                ub.h[0] = *(const bf8v*)(bp);
                ub.h[1] = *(const bf8v*)(bp + 16);
                #pragma unroll
                for (int a = 0; a < 2; ++a) {
                    acc[a][ct] = __builtin_amdgcn_wmma_f32_16x16x32_bf16(
                                     false, Af[a], false, ub.v, (short)0,
                                     acc[a][ct], false, false);
                }
            }
        }
    }

    // ---- relu + store fp32 output ----
    float* obt = out + (size_t)(b * T_ + t) * V_ * O_;
    #pragma unroll
    for (int a = 0; a < 2; ++a) {
        const int vbase = (wave * 2 + a) * 16 + hi * 8;
        #pragma unroll
        for (int ct = 0; ct < 4; ++ct) {
            const int o = ct * 16 + ln;
            #pragma unroll
            for (int r = 0; r < 8; ++r) {
                float val = acc[a][ct][r];
                obt[(size_t)(vbase + r) * O_ + o] = val > 0.f ? val : 0.f;
            }
        }
    }
}

extern "C" void kernel_launch(void* const* d_in, const int* in_sizes, int n_in,
                              void* d_out, int out_size, void* d_ws, size_t ws_size,
                              hipStream_t stream) {
    (void)in_sizes; (void)n_in; (void)out_size; (void)d_ws; (void)ws_size;
    const float* x     = (const float*)d_in[0];
    const float* Att   = (const float*)d_in[1];
    const float* cheb  = (const float*)d_in[2];
    const float* Theta = (const float*)d_in[3];
    float* out = (float*)d_out;

    dim3 grid(B_ * T_);   // one workgroup per (b,t) tile: 2048 blocks
    dim3 block(256);      // 8 waves (wave32)
    hipLaunchKernelGGL(cheb_att_fused, grid, block, 0, stream,
                       x, Att, cheb, Theta, out);
}
